// AdaptiveSoftmax_52553219834145
// MI455X (gfx1250) — matrix operand order repaired and checked by reading, not compiled, not run on hardware
//
#include <hip/hip_runtime.h>
#include <hip/hip_bf16.h>
#include <math.h>

// ---------------------------------------------------------------------------
// Types for CDNA5 WMMA (wave32): V_WMMA_F32_16X16X32_BF16
// ---------------------------------------------------------------------------
typedef __attribute__((ext_vector_type(16))) __bf16 v16bf;
typedef __attribute__((ext_vector_type(8)))  __bf16 v8bf;
typedef __attribute__((ext_vector_type(4)))  __bf16 v4bf;
typedef __attribute__((ext_vector_type(8)))  float  v8f;
typedef __attribute__((ext_vector_type(4)))  float  v4f;

union FragBF16 {
    v16bf v;
    v8bf  h[2];
};

// Problem constants (match the reference)
#define BATCH   4096
#define DIM     1024
#define C0      2000
#define C1      10000
#define C2      50000
#define NHEAD   2002     // C0 + 2
#define NHEADP  2048     // padded head width (multiple of 64)
#define N_T0    8000     // C1 - C0
#define N_T1    40000    // C2 - C1
#define D4      256      // DIM / 4

// ---------------------------------------------------------------------------
// f32 -> bf16 cast, 4-wide vectorized (all tensor sizes are multiples of 4)
// ---------------------------------------------------------------------------
__global__ void cast_f32_bf16_v4(const v4f* __restrict__ src,
                                 v4bf* __restrict__ dst, int n4) {
    int i = blockIdx.x * blockDim.x + threadIdx.x;
    if (i < n4) {
        v4f x = src[i];
        v4bf y;
        y[0] = (__bf16)x[0];
        y[1] = (__bf16)x[1];
        y[2] = (__bf16)x[2];
        y[3] = (__bf16)x[3];
        dst[i] = y;
    }
}

// f32 -> bf16 with row zero-padding (Wh: 2002x1024 -> 2048x1024)
__global__ void cast_pad_f32_bf16(const float* __restrict__ src,
                                  __bf16* __restrict__ dst,
                                  int rows, int padRows, int cols) {
    int i = blockIdx.x * blockDim.x + threadIdx.x;
    int total = padRows * cols;
    if (i < total) {
        int r = i / cols;
        int c = i - r * cols;
        dst[i] = (r < rows) ? (__bf16)src[r * cols + c] : (__bf16)0.0f;
    }
}

// ---------------------------------------------------------------------------
// bf16 WMMA GEMM:  C[M,N] = A[M,K] * W[N,K]^T,   K is a template constant
//   - 256 threads = 8 waves stacked along M, block tile 256(M) x 64(N)
//   - each wave: 32 rows x 64 cols = 8 accumulators
//       per K-step(32): 2 A frags + 4 B frags (shared by both A strips)
//       -> 8 v_wmma per 12 global_load_b128 (1.5 loads / wmma)
//   - two-stage software pipeline (X/Y register sets)
//   - compile-time K folds every fragment offset (s*16*K, strip +16*K, k0)
//     into the 24-bit instruction offset: no per-iteration 64-bit address
//     VALU chains, fewer WMMA<->VALU hazard NOPs.
//   - fragment layouts (bf16), contiguous k = 16*half + s (not the ISA table order):
//       A (16x32): lane<16 holds its row's K[0..15]; lane>=16 K[16..31]
//       B (32x16): lane<16 holds its column's K[0..15]; lane>=16 K[16..31]
// ---------------------------------------------------------------------------
template <int K>
__global__ __launch_bounds__(256)
void gemm_bf16_wmma(const __bf16* __restrict__ A,
                    const __bf16* __restrict__ W,
                    void* __restrict__ Cout,
                    int ldc, int outBf16) {
    const int tid  = threadIdx.x;
    const int wave = tid >> 5;
    const int lane = tid & 31;
    const int half = lane >> 4;   // 0 or 1
    const int l16  = lane & 15;

    const int m0 = blockIdx.y * 256 + wave * 32;   // this wave's 32 rows
    const int n0 = blockIdx.x * 64;                // block's 64 cols

    v8f acc[8] = {};   // [0..3]: rows m0..m0+15 ; [4..7]: rows m0+16..m0+31

    const __bf16* arow = A + (size_t)(m0 + l16) * K;          // strip 0 row
    const __bf16* wcol = W + (size_t)(n0 + l16) * K + half * 16;

    // strip: 0 -> rows m0..m0+15 ; 16*K -> rows m0+16..m0+31 (const offset)
    auto ldA = [&](FragBF16& a, int strip, int k0) {
        a.h[0] = *(const v8bf*)(arow + strip + k0 + half * 16);
        a.h[1] = *(const v8bf*)(arow + strip + k0 + half * 16 + 8);
    };
    auto ldB = [&](FragBF16* b, int k0) {
        #pragma unroll
        for (int s = 0; s < 4; ++s) {
            const __bf16* wp = wcol + s * 16 * K + k0;        // const s*16*K
            b[s].h[0] = *(const v8bf*)(wp);
            b[s].h[1] = *(const v8bf*)(wp + 8);
        }
    };

    FragBF16 aX0, aX1, bX[4], aY0, aY1, bY[4];
    ldA(aX0, 0, 0);
    ldA(aX1, 16 * K, 0);
    ldB(bX, 0);

    #pragma unroll 1
    for (int k0 = 0; k0 < K; k0 += 64) {       // K % 64 == 0 (1024 or 256)
        // prefetch stage Y (k0+32) while stage X computes
        ldA(aY0, 0, k0 + 32);
        ldA(aY1, 16 * K, k0 + 32);
        ldB(bY, k0 + 32);
        #pragma unroll
        for (int s = 0; s < 4; ++s) {
            acc[s] = __builtin_amdgcn_wmma_f32_16x16x32_bf16(
                false, aX0.v, false, bX[s].v, (short)0, acc[s], false, false);
            acc[4 + s] = __builtin_amdgcn_wmma_f32_16x16x32_bf16(
                false, aX1.v, false, bX[s].v, (short)0, acc[4 + s], false, false);
        }

        // prefetch stage X for the next iteration (wrap to 0 on last pass:
        // in-bounds dummy load, keeps the loop branch-free)
        const int k2 = (k0 + 64 < K) ? (k0 + 64) : 0;
        ldA(aX0, 0, k2);
        ldA(aX1, 16 * K, k2);
        ldB(bX, k2);
        #pragma unroll
        for (int s = 0; s < 4; ++s) {
            acc[s] = __builtin_amdgcn_wmma_f32_16x16x32_bf16(
                false, aY0.v, false, bY[s].v, (short)0, acc[s], false, false);
            acc[4 + s] = __builtin_amdgcn_wmma_f32_16x16x32_bf16(
                false, aY1.v, false, bY[s].v, (short)0, acc[4 + s], false, false);
        }
    }

    // D layout: lane<16 -> N=lane, M=r ; lane>=16 -> N=lane-16, M=r+8
    if (outBf16) {
        __bf16* C = (__bf16*)Cout;
        #pragma unroll
        for (int g = 0; g < 2; ++g) {
            const int row_base = m0 + g * 16 + 8 * half;
            #pragma unroll
            for (int s = 0; s < 4; ++s) {
                const int col = n0 + s * 16 + l16;
                #pragma unroll
                for (int r = 0; r < 8; ++r)
                    C[(size_t)(row_base + r) * ldc + col] = (__bf16)acc[4 * g + s][r];
            }
        }
    } else {
        float* C = (float*)Cout;
        #pragma unroll
        for (int g = 0; g < 2; ++g) {
            const int row_base = m0 + g * 16 + 8 * half;
            #pragma unroll
            for (int s = 0; s < 4; ++s) {
                const int col = n0 + s * 16 + l16;
                #pragma unroll
                for (int r = 0; r < 8; ++r)
                    C[(size_t)(row_base + r) * ldc + col] = acc[4 * g + s][r];
            }
        }
    }
}

// ---------------------------------------------------------------------------
// Head log-softmax: one block per row.
//   logits row (padded stride NHEADP, valid cols [0,NHEAD)) + bias
//   writes out[:, 0:C0] and the two cluster log-probs.
// ---------------------------------------------------------------------------
__global__ __launch_bounds__(256)
void head_softmax(const float* __restrict__ hl, const float* __restrict__ bh,
                  float* __restrict__ out, float* __restrict__ cl0,
                  float* __restrict__ cl1) {
    __shared__ float red[256];
    const int row = blockIdx.x;
    const int tid = threadIdx.x;
    const float* rp = hl + (size_t)row * NHEADP;

    float m = -INFINITY;
    for (int c = tid; c < NHEAD; c += 256) m = fmaxf(m, rp[c] + bh[c]);
    red[tid] = m; __syncthreads();
    for (int s = 128; s > 0; s >>= 1) {
        if (tid < s) red[tid] = fmaxf(red[tid], red[tid + s]);
        __syncthreads();
    }
    const float rowmax = red[0]; __syncthreads();

    float acc = 0.0f;
    for (int c = tid; c < NHEAD; c += 256) acc += expf(rp[c] + bh[c] - rowmax);
    red[tid] = acc; __syncthreads();
    for (int s = 128; s > 0; s >>= 1) {
        if (tid < s) red[tid] += red[tid + s];
        __syncthreads();
    }
    const float lse = rowmax + logf(red[0]);

    float* orow = out + (size_t)row * C2;
    for (int c = tid; c < C0; c += 256) orow[c] = rp[c] + bh[c] - lse;
    if (tid == 0) {
        cl0[row] = rp[C0]     + bh[C0]     - lse;
        cl1[row] = rp[C0 + 1] + bh[C0 + 1] - lse;
    }
}

// ---------------------------------------------------------------------------
// Tail fixup: in-place log-softmax over a row slice of d_out plus the
// broadcast cluster log-prob. One block per row; online max/sum-exp with a
// shared-memory tree merge, then a rewrite pass.
// ---------------------------------------------------------------------------
__global__ __launch_bounds__(256)
void tail_fixup(float* __restrict__ base, int ldc, int N,
                const float* __restrict__ cl) {
    __shared__ float sm[256];
    __shared__ float ss[256];
    const int row = blockIdx.x;
    const int tid = threadIdx.x;
    float* p = base + (size_t)row * ldc;

    // per-thread online (m, s);  N >= 8000 so every thread has >= 1 element
    float m = p[tid];
    float s = 1.0f;
    for (int c = tid + 256; c < N; c += 256) {
        float x = p[c];
        if (x <= m) {
            s += expf(x - m);
        } else {
            s = s * expf(m - x) + 1.0f;
            m = x;
        }
    }
    sm[tid] = m; ss[tid] = s; __syncthreads();
    for (int stp = 128; stp > 0; stp >>= 1) {
        if (tid < stp) {
            float m1 = sm[tid], s1 = ss[tid];
            float m2 = sm[tid + stp], s2 = ss[tid + stp];
            float mm = fmaxf(m1, m2);
            sm[tid] = mm;
            ss[tid] = s1 * expf(m1 - mm) + s2 * expf(m2 - mm);
        }
        __syncthreads();
    }
    const float addterm = cl[row] - (sm[0] + logf(ss[0]));
    __syncthreads();

    for (int c = tid; c < N; c += 256) p[c] += addterm;
}

// ---------------------------------------------------------------------------
// Host-side launcher
// ---------------------------------------------------------------------------
extern "C" void kernel_launch(void* const* d_in, const int* in_sizes, int n_in,
                              void* d_out, int out_size, void* d_ws, size_t ws_size,
                              hipStream_t stream) {
    (void)in_sizes; (void)n_in; (void)out_size; (void)ws_size;

    const float* x   = (const float*)d_in[0];   // [4096,1024]
    const float* Wh  = (const float*)d_in[1];   // [2002,1024]
    const float* bh  = (const float*)d_in[2];   // [2002]
    const float* W0a = (const float*)d_in[3];   // [1024,1024]
    const float* W0b = (const float*)d_in[4];   // [8000,1024]
    const float* W1a = (const float*)d_in[5];   // [256,1024]
    const float* W1b = (const float*)d_in[6];   // [40000,256]
    float* out = (float*)d_out;                 // [4096,50000]

    // workspace carve-out (256B aligned)
    char* ws = (char*)d_ws;
    size_t off = 0;
    auto carve = [&](size_t bytes) -> void* {
        void* p = ws + off;
        off = (off + bytes + 255) & ~(size_t)255;
        return p;
    };
    __bf16* xb    = (__bf16*)carve((size_t)BATCH * DIM * 2);
    __bf16* Whb   = (__bf16*)carve((size_t)NHEADP * DIM * 2);
    __bf16* W0ab  = (__bf16*)carve((size_t)DIM * DIM * 2);
    __bf16* W0bb  = (__bf16*)carve((size_t)N_T0 * DIM * 2);
    __bf16* W1ab  = (__bf16*)carve((size_t)D4 * DIM * 2);
    __bf16* W1bb  = (__bf16*)carve((size_t)N_T1 * D4 * 2);
    float*  hlog  = (float*) carve((size_t)BATCH * NHEADP * 4);
    __bf16* h0b   = (__bf16*)carve((size_t)BATCH * DIM * 2);
    __bf16* h1b   = (__bf16*)carve((size_t)BATCH * D4 * 2);
    float*  cl0   = (float*) carve((size_t)BATCH * 4);
    float*  cl1   = (float*) carve((size_t)BATCH * 4);

    auto blk4 = [](size_t n) { return (unsigned)((n / 4 + 255) / 256); };

    // 1) casts to bf16 (vectorized x4)
    cast_f32_bf16_v4<<<blk4((size_t)BATCH * DIM), 256, 0, stream>>>(
        (const v4f*)x, (v4bf*)xb, BATCH * DIM / 4);
    cast_pad_f32_bf16<<<(NHEADP * DIM + 255) / 256, 256, 0, stream>>>(
        Wh, Whb, NHEAD, NHEADP, DIM);
    cast_f32_bf16_v4<<<blk4((size_t)DIM * DIM), 256, 0, stream>>>(
        (const v4f*)W0a, (v4bf*)W0ab, DIM * DIM / 4);
    cast_f32_bf16_v4<<<blk4((size_t)N_T0 * DIM), 256, 0, stream>>>(
        (const v4f*)W0b, (v4bf*)W0bb, N_T0 * DIM / 4);
    cast_f32_bf16_v4<<<blk4((size_t)D4 * DIM), 256, 0, stream>>>(
        (const v4f*)W1a, (v4bf*)W1ab, D4 * DIM / 4);
    cast_f32_bf16_v4<<<blk4((size_t)N_T1 * D4), 256, 0, stream>>>(
        (const v4f*)W1b, (v4bf*)W1bb, N_T1 * D4 / 4);

    // 2) GEMMs (grid = N/64 x M/256, 256 threads, K templated)
    // head logits: [4096,2048] f32 (padded cols are zero-weight -> ignored later)
    gemm_bf16_wmma<DIM><<<dim3(NHEADP / 64, BATCH / 256), 256, 0, stream>>>(
        xb, Whb, hlog, NHEADP, 0);
    // h0 = x @ W0a^T  -> bf16 [4096,1024]
    gemm_bf16_wmma<DIM><<<dim3(DIM / 64, BATCH / 256), 256, 0, stream>>>(
        xb, W0ab, h0b, DIM, 1);
    // h1 = x @ W1a^T  -> bf16 [4096,256]
    gemm_bf16_wmma<DIM><<<dim3(D4 / 64, BATCH / 256), 256, 0, stream>>>(
        xb, W1ab, h1b, D4, 1);
    // t0 logits straight into d_out[:, 2000:10000]
    gemm_bf16_wmma<DIM><<<dim3(N_T0 / 64, BATCH / 256), 256, 0, stream>>>(
        h0b, W0bb, out + C0, C2, 0);
    // t1 logits straight into d_out[:, 10000:50000]   (K = 256)
    gemm_bf16_wmma<D4><<<dim3(N_T1 / 64, BATCH / 256), 256, 0, stream>>>(
        h1b, W1bb, out + C1, C2, 0);

    // 3) head log-softmax -> out[:, :2000] + cluster log-probs
    head_softmax<<<BATCH, 256, 0, stream>>>(hlog, bh, out, cl0, cl1);

    // 4) in-place tail log-softmax + cluster broadcast
    tail_fixup<<<BATCH, 256, 0, stream>>>(out + C0, C2, N_T0, cl0);
    tail_fixup<<<BATCH, 256, 0, stream>>>(out + C1, C2, N_T1, cl1);
}
